// SelfAttention_57114475102541
// MI455X (gfx1250) — matrix operand order repaired and checked by reading, not compiled, not run on hardware
//
#include <hip/hip_runtime.h>

typedef __attribute__((ext_vector_type(16))) __bf16          v16bf;
typedef __attribute__((ext_vector_type(8)))  float           v8f;
typedef __attribute__((ext_vector_type(8)))  unsigned short  v8u;
typedef __attribute__((ext_vector_type(16))) unsigned short  v16u;
typedef __attribute__((ext_vector_type(4)))  unsigned int    u32x4;
typedef __attribute__((ext_vector_type(8)))  unsigned int    u32x8;

#define D_EMBED 512
#define N_HEADS 8
#define D_HEAD  64
#define SEQ     4096
#define BATCH   2
#define QKV_N   (3 * D_EMBED)

#ifndef USE_TDM
#define USE_TDM 1            // Tensor Data Mover async staging (flip to 0 on asm failure)
#endif

// ---------- helpers ----------

static __device__ __forceinline__ unsigned short f2bf(float f) {
  unsigned u = __float_as_uint(f);
  u += 0x7FFFu + ((u >> 16) & 1u);          // round-to-nearest-even
  return (unsigned short)(u >> 16);
}

static __device__ __forceinline__ v16bf frag_cast(v8u lo, v8u hi) {
  v16u t = __builtin_shufflevector(lo, hi, 0,1,2,3,4,5,6,7,8,9,10,11,12,13,14,15);
  return __builtin_bit_cast(v16bf, t);
}

// A-fragment (16x32 bf16), ISA 7.12.2: lane l -> m=l&15, khalf=(l<16)?0:8;
// halves 0-7 = [m][kbase+khalf+0..7], halves 8-15 = [m][kbase+khalf+16..23]
static __device__ __forceinline__ v16bf load_afrag(const unsigned short* base,
                                                   int rowStride, int m,
                                                   int kbase, int lane) {
  int khalf = (lane < 16) ? 0 : 8;
  const v8u* p0 = (const v8u*)(base + (size_t)m * rowStride + kbase + khalf);
  const v8u* p1 = (const v8u*)(base + (size_t)m * rowStride + kbase + khalf + 16);
  return frag_cast(*p0, *p1);
}

// B-fragment (32x16 bf16): lane l -> n=l&15, khalf=(l<16)?0:8;
// halves 0-7 = [kbase+khalf+0..7][n], halves 8-15 = [kbase+khalf+16..23][n]
static __device__ __forceinline__ v16bf load_bfrag(const unsigned short* base,
                                                   int rowStride, int krow, int nbase) {
  const int l = krow & 31, kb = krow & ~31;
  const unsigned short* p = base + nbase + (l & 15) + (size_t)(kb + ((l >> 4) << 3)) * rowStride;
  v16u t;
#pragma unroll
  for (int s = 0; s < 8; ++s) {
    t[s]     = p[(size_t)s * rowStride];
    t[s + 8] = p[(size_t)(16 + s) * rowStride];
  }
  return __builtin_bit_cast(v16bf, t);
}

static __device__ __forceinline__ v8f wmma_bf16(v16bf a, v16bf b, v8f c) {
  return __builtin_amdgcn_wmma_f32_16x16x32_bf16(false, a, false, b,
                                                 (short)0, c, false, false);
}

static __device__ __forceinline__ v8f zero8() {
  v8f z = {0.f, 0.f, 0.f, 0.f, 0.f, 0.f, 0.f, 0.f};
  return z;
}

static __device__ __forceinline__ unsigned ldsoff(const void* p) {
  return (unsigned)(unsigned long long)p;   // low 32 bits of LDS aperture addr = LDS offset
}

#if USE_TDM
// Issue a 2D TDM tile load: tile_h rows of tile_w bf16 elems, global row
// stride row_stride_elems, packed dense into LDS at lds_off. Groups 2/3 zero
// (dims 3/4 unused). Tracked with TENSORcnt.
static __device__ __forceinline__ void tdm_load_2d(unsigned lds_off, const void* gptr,
                                                   unsigned tile_w, unsigned tile_h,
                                                   unsigned row_stride_elems) {
  unsigned long long ga = (unsigned long long)gptr;
  u32x4 g0 = { 1u,                                   // count=1, user descriptor
               lds_off,                              // lds_addr
               (unsigned)ga,                         // global_addr[31:0]
               (unsigned)(ga >> 32) | 0x80000000u }; // global_addr[56:32] | type=2
  u32x8 g1 = { 1u << 16,                            // data_size=2B
               tile_w << 16,                        // tensor_dim0[15:0]
               (tile_w >> 16) | (tile_h << 16),     // tensor_dim0 hi | tensor_dim1 lo
               (tile_h >> 16) | (tile_w << 16),     // tensor_dim1 hi | tile_dim0
               tile_h,                              // tile_dim1 (tile_dim2 = 0)
               row_stride_elems,                    // tensor_dim0_stride[31:0]
               0u, 0u };
  u32x4 gz = { 0u, 0u, 0u, 0u };
  asm volatile("tensor_load_to_lds %0, %1, %2, %3"
               :: "s"(g0), "s"(g1), "s"(gz), "s"(gz)
               : "memory");
}
#endif

// ---------- f32 -> bf16 conversion ----------

__global__ void cvt_f32_bf16(const float* __restrict__ in,
                             unsigned short* __restrict__ out, int n) {
  int i = blockIdx.x * blockDim.x + threadIdx.x;
  int stride = gridDim.x * blockDim.x;
  for (; i < n; i += stride) out[i] = f2bf(in[i]);
}

// ---------- fused convert + transpose: W[N,K] f32 -> Wt[K,N] bf16 ----------
// grid (N/128, K/64); tiled through LDS so both sides stay coalesced.

__global__ __launch_bounds__(256)
void cvt_transpose(const float* __restrict__ W, unsigned short* __restrict__ Wt,
                   int N, int K) {
  __shared__ unsigned short t[64 * 130];
  const int tid = threadIdx.x;
  const int n0 = blockIdx.x * 128, k0 = blockIdx.y * 64;
  for (int i = tid; i < 128 * 64; i += 256) {
    int k = i & 63, n = i >> 6;                 // consecutive tid -> consecutive k
    t[k * 130 + n] = f2bf(W[(size_t)(n0 + n) * K + k0 + k]);
  }
  __syncthreads();
  for (int i = tid; i < 64 * 128; i += 256) {
    int n = i & 127, k = i >> 7;                // consecutive tid -> consecutive n
    Wt[(size_t)(k0 + k) * N + n0 + n] = t[k * 130 + n];
  }
}

// ---------- transpose K section of qkv -> ktb[b*H+h][d][s] (bf16) ----------
// grid (SEQ/128, B*N_HEADS)

__global__ __launch_bounds__(256)
void transpose_k(const unsigned short* __restrict__ qkv,
                 unsigned short* __restrict__ ktb) {
  __shared__ unsigned short t[64 * 130];
  const int tid = threadIdx.x;
  const int s0 = blockIdx.x * 128;
  const int bh = blockIdx.y, b = bh >> 3, h = bh & 7;
  for (int i = tid; i < 64 * 128; i += 256) {
    int d = i & 63, s = i >> 6;                 // consecutive tid -> consecutive d
    t[d * 130 + s] =
        qkv[((size_t)b * SEQ + s0 + s) * QKV_N + D_EMBED + h * D_HEAD + d];
  }
  __syncthreads();
  for (int i = tid; i < 64 * 128; i += 256) {
    int s = i & 127, d = i >> 7;                // consecutive tid -> consecutive s
    ktb[((size_t)bh * 64 + d) * SEQ + s0 + s] = t[d * 130 + s];
  }
}

// ---------- bf16 WMMA GEMM: out[M,N] = A[M,K] @ Wt[K,N] + bias ----------
// BM=128, BN=128, BK=64, 256 threads (8 waves: 4 row x 2 col, 32x64 per wave).
// Double-buffered LDS; tiles staged by the TDM (wave 0) overlapped with compute.

__global__ __launch_bounds__(256)
void gemm_bf16_kernel(const unsigned short* __restrict__ A,
                      const unsigned short* __restrict__ Wt,
                      const float* __restrict__ bias,
                      unsigned short* __restrict__ outb,
                      float* __restrict__ outf,
                      int M, int N, int K, int write_f32) {
  __shared__ __attribute__((aligned(32))) unsigned short As[2][128 * 64];
  __shared__ __attribute__((aligned(32))) unsigned short Bs[2][64 * 128];

  const int tid  = threadIdx.x;
  const int lane = tid & 31;
  const int wid  = tid >> 5;
  const int wr   = wid & 3;
  const int wc   = wid >> 2;
  const int n0   = blockIdx.x * 128;
  const int m0   = blockIdx.y * 128;

  v8f acc[2][4];
#pragma unroll
  for (int i = 0; i < 2; ++i)
#pragma unroll
    for (int j = 0; j < 4; ++j) acc[i][j] = zero8();

  const unsigned short* abase = A + (size_t)m0 * K;   // + kt*64
  const unsigned short* bbase = Wt + n0;              // + (kt*64)*N
  const int steps = K >> 6;

#if USE_TDM
  if (wid == 0) {
    tdm_load_2d(ldsoff(As[0]), abase, 64, 128, (unsigned)K);
    tdm_load_2d(ldsoff(Bs[0]), bbase, 128, 64, (unsigned)N);
    __builtin_amdgcn_s_wait_tensorcnt(0);
  }
  __syncthreads();
#endif

  for (int kt = 0; kt < steps; ++kt) {
    const int cur = kt & 1;
#if USE_TDM
    if (wid == 0 && kt + 1 < steps) {   // prefetch next tiles into alternate buffers
      tdm_load_2d(ldsoff(As[1 - cur]), abase + (kt + 1) * 64, 64, 128, (unsigned)K);
      tdm_load_2d(ldsoff(Bs[1 - cur]), bbase + (size_t)(kt + 1) * 64 * N, 128, 64, (unsigned)N);
    }
#else
    __syncthreads();
    {
      int r = tid >> 1, c = (tid & 1) * 32;           // A: 128x64
      const unsigned short* ga = abase + (size_t)r * K + kt * 64 + c;
      unsigned short* da = As[cur] + r * 64 + c;
#pragma unroll
      for (int q = 0; q < 4; ++q) *(v8u*)(da + q * 8) = *(const v8u*)(ga + q * 8);
      int rb = tid >> 2, cb = (tid & 3) * 32;         // B: 64x128
      const unsigned short* gb = bbase + (size_t)(kt * 64 + rb) * N + cb;
      unsigned short* db = Bs[cur] + rb * 128 + cb;
#pragma unroll
      for (int q = 0; q < 4; ++q) *(v8u*)(db + q * 8) = *(const v8u*)(gb + q * 8);
    }
    __syncthreads();
#endif

    v16bf af[2][2], bfg[4][2];
#pragma unroll
    for (int m16 = 0; m16 < 2; ++m16)
#pragma unroll
      for (int j = 0; j < 2; ++j)
        af[m16][j] = load_afrag(As[cur] + (wr * 32 + m16 * 16) * 64, 64,
                                lane & 15, j * 32, lane);
#pragma unroll
    for (int nt = 0; nt < 4; ++nt)
#pragma unroll
      for (int j = 0; j < 2; ++j)
        bfg[nt][j] = load_bfrag(Bs[cur], 128, j * 32 + lane, wc * 64 + nt * 16);
#pragma unroll
    for (int m16 = 0; m16 < 2; ++m16)
#pragma unroll
      for (int nt = 0; nt < 4; ++nt)
#pragma unroll
        for (int j = 0; j < 2; ++j)
          acc[m16][nt] = wmma_bf16(af[m16][j], bfg[nt][j], acc[m16][nt]);

#if USE_TDM
    if (wid == 0) __builtin_amdgcn_s_wait_tensorcnt(0);
    __syncthreads();
#endif
  }

  // epilogue: C layout (VGPR r, lane) -> row r+8*(lane>=16), col lane&15
  const int rowoff = (lane < 16) ? 0 : 8;
  const int nlan   = lane & 15;
#pragma unroll
  for (int m16 = 0; m16 < 2; ++m16)
#pragma unroll
    for (int nt = 0; nt < 4; ++nt)
#pragma unroll
      for (int r = 0; r < 8; ++r) {
        int row = m0 + wr * 32 + m16 * 16 + r + rowoff;
        int col = n0 + wc * 64 + nt * 16 + nlan;
        float v = acc[m16][nt][r] + bias[col];
        if (write_f32) outf[(size_t)row * N + col] = v;
        else           outb[(size_t)row * N + col] = f2bf(v);
      }
}

// ---------- fused flash attention (bf16 WMMA, f32 online softmax) ----------
// grid = (SEQ/128, N_HEADS, BATCH), 256 threads; wave w owns Q rows w*16..+15.
// K^T and V 64x64 tiles double-buffered, staged by the TDM.

__global__ __launch_bounds__(256)
void attn_kernel(const unsigned short* __restrict__ qkv,
                 const unsigned short* __restrict__ ktb,
                 unsigned short* __restrict__ obuf) {
  __shared__ __attribute__((aligned(32))) unsigned short Kt[2][64 * 64];
  __shared__ __attribute__((aligned(32))) unsigned short Vs[2][64 * 64];
  __shared__ __attribute__((aligned(32))) unsigned short Pb[8 * 16 * 64];

  const int tid  = threadIdx.x;
  const int lane = tid & 31;
  const int wid  = tid >> 5;
  const int h    = blockIdx.y;
  const int b    = blockIdx.z;
  const int q0   = blockIdx.x * 128;
  const size_t seqbase = (size_t)b * SEQ;

  const unsigned short* qbase =
      qkv + (seqbase + q0 + wid * 16) * QKV_N + h * D_HEAD;
  v16bf qf[2];
  qf[0] = load_afrag(qbase, QKV_N, lane & 15, 0,  lane);
  qf[1] = load_afrag(qbase, QKV_N, lane & 15, 32, lane);

  const unsigned short* ktbase = ktb + (size_t)(b * N_HEADS + h) * 64 * SEQ;
  const unsigned short* vbase  = qkv + seqbase * QKV_N + 2 * D_EMBED + h * D_HEAD;

  const float scale = 0.125f;           // 1/sqrt(64)
  float mrow[8], lrow[8];
  v8f o[4];
#pragma unroll
  for (int r = 0; r < 8; ++r) { mrow[r] = -1e30f; lrow[r] = 0.f; }
#pragma unroll
  for (int dt = 0; dt < 4; ++dt) o[dt] = zero8();

  unsigned short* pw = Pb + wid * (16 * 64);
  const int rowoff = (lane < 16) ? 0 : 8;
  const int nlan   = lane & 15;
  const int NKB    = SEQ / 64;

#if USE_TDM
  if (wid == 0) {
    tdm_load_2d(ldsoff(Kt[0]), ktbase, 64, 64, SEQ);
    tdm_load_2d(ldsoff(Vs[0]), vbase, 64, 64, QKV_N);
    __builtin_amdgcn_s_wait_tensorcnt(0);
  }
  __syncthreads();
#endif

  for (int kb = 0; kb < NKB; ++kb) {
    const int cur = kb & 1;
    const int kv0 = kb * 64;
#if USE_TDM
    if (wid == 0 && kb + 1 < NKB) {     // prefetch next K^T / V tiles
      tdm_load_2d(ldsoff(Kt[1 - cur]), ktbase + kv0 + 64, 64, 64, SEQ);
      tdm_load_2d(ldsoff(Vs[1 - cur]), vbase + (size_t)(kv0 + 64) * QKV_N, 64, 64, QKV_N);
    }
#else
    __syncthreads();
    {
      int r = tid >> 2, c = (tid & 3) * 16;
      const unsigned short* gk = ktbase + (size_t)r * SEQ + kv0 + c;
      *(v8u*)(Kt[cur] + r * 64 + c)     = *(const v8u*)(gk);
      *(v8u*)(Kt[cur] + r * 64 + c + 8) = *(const v8u*)(gk + 8);
      const unsigned short* gv = vbase + (size_t)(kv0 + r) * QKV_N + c;
      *(v8u*)(Vs[cur] + r * 64 + c)     = *(const v8u*)(gv);
      *(v8u*)(Vs[cur] + r * 64 + c + 8) = *(const v8u*)(gv + 8);
    }
    __syncthreads();
#endif
    const unsigned short* KtB = Kt[cur];
    const unsigned short* VsB = Vs[cur];

    // scores: S = (Q @ K^T) * scale -> 8 WMMAs
    v8f s[4];
#pragma unroll
    for (int nt = 0; nt < 4; ++nt) {
      v8f acc = zero8();
#pragma unroll
      for (int j = 0; j < 2; ++j)
        acc = wmma_bf16(qf[j], load_bfrag(KtB, 64, j * 32 + lane, nt * 16), acc);
#pragma unroll
      for (int r = 0; r < 8; ++r) s[nt][r] = acc[r] * scale;
    }

    // online softmax (row = (r, lane-half); N over the 16 lanes of the half)
    float mnew[8];
#pragma unroll
    for (int r = 0; r < 8; ++r) {
      mnew[r] = mrow[r];
#pragma unroll
      for (int nt = 0; nt < 4; ++nt) mnew[r] = fmaxf(mnew[r], s[nt][r]);
    }
#pragma unroll
    for (int mask = 8; mask >= 1; mask >>= 1)
#pragma unroll
      for (int r = 0; r < 8; ++r)
        mnew[r] = fmaxf(mnew[r], __shfl_xor(mnew[r], mask, 32));

    float rs[8];
#pragma unroll
    for (int r = 0; r < 8; ++r) {
      float alpha = __expf(mrow[r] - mnew[r]);
      mrow[r] = mnew[r];
      lrow[r] *= alpha;
      rs[r] = 0.f;
#pragma unroll
      for (int dt = 0; dt < 4; ++dt) o[dt][r] *= alpha;
    }
#pragma unroll
    for (int nt = 0; nt < 4; ++nt)
#pragma unroll
      for (int r = 0; r < 8; ++r) {
        float p = __expf(s[nt][r] - mrow[r]);
        s[nt][r] = p;
        rs[r] += p;
      }
#pragma unroll
    for (int mask = 8; mask >= 1; mask >>= 1)
#pragma unroll
      for (int r = 0; r < 8; ++r) rs[r] += __shfl_xor(rs[r], mask, 32);
#pragma unroll
    for (int r = 0; r < 8; ++r) lrow[r] += rs[r];

    // repack P (C layout) -> bf16 row-major 16x64 in wave-private LDS;
    // same-wave LDS ops are in-order (DScnt), so following loads see the data
#pragma unroll
    for (int nt = 0; nt < 4; ++nt)
#pragma unroll
      for (int r = 0; r < 8; ++r)
        pw[(r + rowoff) * 64 + nt * 16 + nlan] = f2bf(s[nt][r]);

    // O += P @ V -> 8 WMMAs
    v16bf pf[2];
    pf[0] = load_afrag(pw, 64, lane & 15, 0,  lane);
    pf[1] = load_afrag(pw, 64, lane & 15, 32, lane);
#pragma unroll
    for (int dt = 0; dt < 4; ++dt)
#pragma unroll
      for (int j = 0; j < 2; ++j)
        o[dt] = wmma_bf16(pf[j], load_bfrag(VsB, 64, j * 32 + lane, dt * 16), o[dt]);

#if USE_TDM
    if (wid == 0) __builtin_amdgcn_s_wait_tensorcnt(0);
    __syncthreads();
#endif
  }

  // epilogue: O / l -> bf16 [B*S, D] head-interleaved buffer
  unsigned short* orow =
      obuf + (seqbase + q0 + wid * 16) * D_EMBED + h * D_HEAD;
#pragma unroll
  for (int r = 0; r < 8; ++r) {
    float inv = 1.f / lrow[r];
#pragma unroll
    for (int dt = 0; dt < 4; ++dt)
      orow[(size_t)(r + rowoff) * D_EMBED + dt * 16 + nlan] = f2bf(o[dt][r] * inv);
  }
}

// ---------- launcher ----------

extern "C" void kernel_launch(void* const* d_in, const int* in_sizes, int n_in,
                              void* d_out, int out_size, void* d_ws, size_t ws_size,
                              hipStream_t stream) {
  const float* x    = (const float*)d_in[0];
  const float* wqkv = (const float*)d_in[1];
  const float* bqkv = (const float*)d_in[2];
  const float* wout = (const float*)d_in[3];
  const float* bout = (const float*)d_in[4];
  float* out = (float*)d_out;
  (void)in_sizes; (void)n_in; (void)out_size; (void)ws_size;

  char* ws = (char*)d_ws;
  size_t off = 0;
  auto carve = [&](size_t bytes) -> char* {
    char* p = ws + off;
    off = (off + bytes + 255) & ~(size_t)255;
    return p;
  };
  const size_t MTOK = (size_t)BATCH * SEQ;  // 8192 tokens
  unsigned short* xb   = (unsigned short*)carve(MTOK * D_EMBED * 2);
  unsigned short* wqt  = (unsigned short*)carve((size_t)D_EMBED * QKV_N * 2);   // [K=512][N=1536]
  unsigned short* wot  = (unsigned short*)carve((size_t)D_EMBED * D_EMBED * 2); // [K=512][N=512]
  unsigned short* qkvb = (unsigned short*)carve(MTOK * QKV_N * 2);
  unsigned short* ktb  = (unsigned short*)carve((size_t)BATCH * N_HEADS * D_HEAD * SEQ * 2);
  unsigned short* ob   = (unsigned short*)carve(MTOK * D_EMBED * 2);

  cvt_f32_bf16<<<2048, 256, 0, stream>>>(x, xb, (int)(MTOK * D_EMBED));
  cvt_transpose<<<dim3(QKV_N / 128, D_EMBED / 64), 256, 0, stream>>>(wqkv, wqt, QKV_N, D_EMBED);
  cvt_transpose<<<dim3(D_EMBED / 128, D_EMBED / 64), 256, 0, stream>>>(wout, wot, D_EMBED, D_EMBED);

  dim3 g1(QKV_N / 128, (unsigned)(MTOK / 128));
  gemm_bf16_kernel<<<g1, 256, 0, stream>>>(xb, wqt, bqkv, qkvb, nullptr,
                                           (int)MTOK, QKV_N, D_EMBED, 0);

  transpose_k<<<dim3(SEQ / 128, BATCH * N_HEADS), 256, 0, stream>>>(qkvb, ktb);

  dim3 ga(SEQ / 128, N_HEADS, BATCH);
  attn_kernel<<<ga, 256, 0, stream>>>(qkvb, ktb, ob);

  dim3 g2(D_EMBED / 128, (unsigned)(MTOK / 128));
  gemm_bf16_kernel<<<g2, 256, 0, stream>>>(ob, wot, bout, nullptr, out,
                                           (int)MTOK, D_EMBED, D_EMBED, 1);
}
